// USSPA_76175539962318
// MI455X (gfx1250) — matrix operand order repaired and checked by reading, not compiled, not run on hardware
//
#include <hip/hip_runtime.h>
#include <hip/hip_bf16.h>
#include <math.h>

// USSPA point-completion network on MI455X (gfx1250, wave32).
// Core GEMM path: v_wmma_f32_16x16x32_f16, f32 accumulation, 64x16 tile/wave,
// A-fragments for all 4 subtiles loaded ahead of the 4 back-to-back WMMAs.
// FPS: 1 workgroup (1024 thr) / batch, points + min-dist in LDS, wave-shuffle
// argmax reduction (2 barriers per step).

typedef __attribute__((ext_vector_type(16))) _Float16 v16h;
typedef __attribute__((ext_vector_type(8)))  float    v8f;

#define THREADS 256
#define FPS_THREADS 1024

// ---- fragment loaders -----------------------------------------------------
__device__ __forceinline__ v16h pack16(float4 a, float4 b, float4 c, float4 d)
{
  v16h r;
  r[0]  = (_Float16)a.x; r[1]  = (_Float16)a.y; r[2]  = (_Float16)a.z; r[3]  = (_Float16)a.w;
  r[4]  = (_Float16)b.x; r[5]  = (_Float16)b.y; r[6]  = (_Float16)b.z; r[7]  = (_Float16)b.w;
  r[8]  = (_Float16)c.x; r[9]  = (_Float16)c.y; r[10] = (_Float16)c.z; r[11] = (_Float16)c.w;
  r[12] = (_Float16)d.x; r[13] = (_Float16)d.y; r[14] = (_Float16)d.z; r[15] = (_Float16)d.w;
  return r;
}
// A 16x32 f16 (ISA 7.12.2): lane half h reads K = h*8..h*8+7 and 16+h*8..+7
__device__ __forceinline__ v16h load_a32(const float* __restrict__ p, int half)
{
  const float* q = p + half * 8;
  return pack16(*(const float4*)q, *(const float4*)(q + 4),
                *(const float4*)(q + 16), *(const float4*)(q + 20));
}
// B 32x16 f16: lane half h reads K = h*8..h*8+7 and 16+h*8..+7 (column = lane&15)
__device__ __forceinline__ v16h load_b32(const float* __restrict__ p, int half)
{
  const float* q = p + half * 8;
  return pack16(*(const float4*)q, *(const float4*)(q + 4),
                *(const float4*)(q + 16), *(const float4*)(q + 20));
}
// tail (K % 32): clamp index, mask with cndmask (branchless)
__device__ __forceinline__ v16h load_a_tail(const float* __restrict__ Xr,
                                            int k0, int K, int half)
{
  v16h a;
#pragma unroll
  for (int v = 0; v < 8; ++v) {
    int ka = k0 + ((v < 4) ? (half * 8 + 2 * v) : (16 + half * 8 + 2 * (v - 4)));
    float x0 = Xr[ka     < K ? ka     : K - 1];
    float x1 = Xr[ka + 1 < K ? ka + 1 : K - 1];
    a[2 * v]     = (_Float16)(ka     < K ? x0 : 0.f);
    a[2 * v + 1] = (_Float16)(ka + 1 < K ? x1 : 0.f);
  }
  return a;
}
__device__ __forceinline__ v16h load_b_tail(const float* __restrict__ Wr,
                                            int k0, int K, int half)
{
  v16h b;
#pragma unroll
  for (int v = 0; v < 8; ++v) {
    int kb = k0 + ((v < 4) ? (half * 8 + 2 * v) : (16 + half * 8 + 2 * (v - 4)));
    float x0 = Wr[kb     < K ? kb     : K - 1];
    float x1 = Wr[kb + 1 < K ? kb + 1 : K - 1];
    b[2 * v]     = (_Float16)(kb     < K ? x0 : 0.f);
    b[2 * v + 1] = (_Float16)(kb + 1 < K ? x1 : 0.f);
  }
  return b;
}

__device__ __forceinline__ float apply_act(float v, int act)
{
  if (act == 1)      v = fmaxf(v, 0.f);
  else if (act == 2) v = tanhf(v);
  else if (act == 3) v = 1.f / (1.f + expf(-v));
  return v;
}

// ---------------------------------------------------------------------------
// WMMA GEMM: Y[M,Co] = act(X[M,K] * W[Co,K]^T + bias).
// k_gemm4: requires M % 64 == 0 -> every active wave owns a full 64x16 tile.
// All four A fragments are materialized before the 4 WMMAs: distinct VGPR
// sets, no WMMA->VALU WAR hazard NOPs between subtiles.
// ---------------------------------------------------------------------------
__global__ __launch_bounds__(THREADS) void k_gemm4(
    const float* __restrict__ X, const float* __restrict__ W,
    const float* __restrict__ bias, float* __restrict__ Y,
    int M, int K, int Co, int act, int ldx, int ldw)
{
  const int lane = threadIdx.x & 31;
  const int wave = threadIdx.x >> 5;
  const int m_base = (blockIdx.x * 8 + wave) * 64;
  if (m_base >= M) return;

  const int half = lane >> 4;
  const int r16  = lane & 15;
  const int bcol = blockIdx.y * 16 + r16;
  const bool bcol_ok = bcol < Co;
  const int bc = bcol_ok ? bcol : (Co - 1);   // clamp: pollutes only columns
                                              // that are never stored
  const float* A0 = X + (size_t)(m_base + r16) * ldx;
  const float* Wr = W + (size_t)bc * ldw;

  v8f acc0 = {}, acc1 = {}, acc2 = {}, acc3 = {};
  const int kfull = K & ~31;
  for (int k0 = 0; k0 < kfull; k0 += 32) {
    if (k0 + 32 < kfull) {                    // global_prefetch_b8 next slab
      __builtin_prefetch(A0 + k0 + 32, 0, 1);
      __builtin_prefetch(Wr + k0 + 32, 0, 1);
    }
    v16h b  = load_b32(Wr + k0, half);
    v16h a0 = load_a32(A0 + k0, half);
    v16h a1 = load_a32(A0 + (size_t)16 * ldx + k0, half);
    v16h a2 = load_a32(A0 + (size_t)32 * ldx + k0, half);
    v16h a3 = load_a32(A0 + (size_t)48 * ldx + k0, half);
    acc0 = __builtin_amdgcn_wmma_f32_16x16x32_f16(false, a0, false, b, (short)0, acc0, false, false);
    acc1 = __builtin_amdgcn_wmma_f32_16x16x32_f16(false, a1, false, b, (short)0, acc1, false, false);
    acc2 = __builtin_amdgcn_wmma_f32_16x16x32_f16(false, a2, false, b, (short)0, acc2, false, false);
    acc3 = __builtin_amdgcn_wmma_f32_16x16x32_f16(false, a3, false, b, (short)0, acc3, false, false);
  }
  if (kfull < K) {
    v16h b  = load_b_tail(Wr, kfull, K, half);
    v16h a0 = load_a_tail(A0, kfull, K, half);
    v16h a1 = load_a_tail(A0 + (size_t)16 * ldx, kfull, K, half);
    v16h a2 = load_a_tail(A0 + (size_t)32 * ldx, kfull, K, half);
    v16h a3 = load_a_tail(A0 + (size_t)48 * ldx, kfull, K, half);
    acc0 = __builtin_amdgcn_wmma_f32_16x16x32_f16(false, a0, false, b, (short)0, acc0, false, false);
    acc1 = __builtin_amdgcn_wmma_f32_16x16x32_f16(false, a1, false, b, (short)0, acc1, false, false);
    acc2 = __builtin_amdgcn_wmma_f32_16x16x32_f16(false, a2, false, b, (short)0, acc2, false, false);
    acc3 = __builtin_amdgcn_wmma_f32_16x16x32_f16(false, a3, false, b, (short)0, acc3, false, false);
  }

  if (!bcol_ok) return;
  const float bv = bias[bcol];
  v8f accs[4] = {acc0, acc1, acc2, acc3};
#pragma unroll
  for (int t = 0; t < 4; ++t)
#pragma unroll
    for (int r = 0; r < 8; ++r) {             // C/D: VGPR r -> M = r / r+8
      int row = m_base + 16 * t + r + half * 8;
      Y[(size_t)row * Co + bcol] = apply_act(accs[t][r] + bv, act);
    }
}

// Generic variant (M % 16 == 0 only); used for tiny head GEMMs (M = 16/32/48).
// nsub forced into an SGPR so guards lower to scalar branches.
__global__ __launch_bounds__(THREADS) void k_gemm(
    const float* __restrict__ X, const float* __restrict__ W,
    const float* __restrict__ bias, float* __restrict__ Y,
    int M, int K, int Co, int act, int ldx, int ldw)
{
  const int lane = threadIdx.x & 31;
  const int wave = threadIdx.x >> 5;
  const int m_base = (blockIdx.x * 8 + wave) * 64;
  if (m_base >= M) return;
  int nsub = (M - m_base) >> 4;
  if (nsub > 4) nsub = 4;
  nsub = __builtin_amdgcn_readfirstlane(nsub);   // wave-uniform -> s_cbranch

  const int half = lane >> 4;
  const int r16  = lane & 15;
  const int bcol = blockIdx.y * 16 + r16;
  const bool bcol_ok = bcol < Co;
  const int bc = bcol_ok ? bcol : (Co - 1);
  const float* A0 = X + (size_t)(m_base + r16) * ldx;
  const float* Wr = W + (size_t)bc * ldw;

  v8f acc[4] = {{}, {}, {}, {}};
  const int kfull = K & ~31;
  for (int k0 = 0; k0 < kfull; k0 += 32) {
    v16h b = load_b32(Wr + k0, half);
#pragma unroll
    for (int t = 0; t < 4; ++t) {
      if (t < nsub) {
        v16h a = load_a32(A0 + (size_t)(16 * t) * ldx + k0, half);
        acc[t] = __builtin_amdgcn_wmma_f32_16x16x32_f16(false, a, false, b,
                                                        (short)0, acc[t],
                                                        false, false);
      }
    }
  }
  if (kfull < K) {
    v16h b = load_b_tail(Wr, kfull, K, half);
#pragma unroll
    for (int t = 0; t < 4; ++t) {
      if (t < nsub) {
        v16h a = load_a_tail(A0 + (size_t)(16 * t) * ldx, kfull, K, half);
        acc[t] = __builtin_amdgcn_wmma_f32_16x16x32_f16(false, a, false, b,
                                                        (short)0, acc[t],
                                                        false, false);
      }
    }
  }

  if (!bcol_ok) return;
  const float bv = bias[bcol];
#pragma unroll
  for (int t = 0; t < 4; ++t) {
    if (t < nsub) {
#pragma unroll
      for (int r = 0; r < 8; ++r) {
        int row = m_base + 16 * t + r + half * 8;
        Y[(size_t)row * Co + bcol] = apply_act(acc[t][r] + bv, act);
      }
    }
  }
}

// ---------------------------------------------------------------------------
// g[b,c] = max_n X[b*Np+n, c]
// ---------------------------------------------------------------------------
__global__ void k_rowmax(const float* __restrict__ X, float* __restrict__ g,
                         int Np, int C)
{
  int b = blockIdx.x;
  int c = blockIdx.y * blockDim.x + threadIdx.x;
  if (c >= C) return;
  const float* p = X + (size_t)b * Np * C + c;
  float m = -3.4e38f;
  for (int n = 0; n < Np; ++n) m = fmaxf(m, p[(size_t)n * C]);
  g[(size_t)b * C + c] = m;
}

// out[row] = [f[row] (C) | g[row/Np] (C)]
__global__ void k_concat_fg(const float* __restrict__ f, const float* __restrict__ g,
                            float* __restrict__ o, int rows, int C, int Np)
{
  long i = (long)blockIdx.x * blockDim.x + threadIdx.x;
  long total = (long)rows * 2 * C;
  if (i >= total) return;
  int W2 = 2 * C;
  int row = (int)(i / W2), c = (int)(i % W2);
  o[i] = (c < C) ? f[(size_t)row * C + c] : g[(size_t)(row / Np) * C + (c - C)];
}

// out[row] = [a[row] (Ca) | b[row] (Cb)]
__global__ void k_concat2(const float* __restrict__ a, const float* __restrict__ b,
                          float* __restrict__ o, int rows, int Ca, int Cb)
{
  long i = (long)blockIdx.x * blockDim.x + threadIdx.x;
  int Wt = Ca + Cb;
  long total = (long)rows * Wt;
  if (i >= total) return;
  int row = (int)(i / Wt), c = (int)(i % Wt);
  o[i] = (c < Ca) ? a[(size_t)row * Ca + c] : b[(size_t)row * Cb + (c - Ca)];
}

// out[row*ldo + c] = [p (3) | x (Cx) | g[row/Np] (Cg)]  (padded row stride)
__global__ void k_concat_pxg(const float* __restrict__ p, const float* __restrict__ x,
                             const float* __restrict__ g, float* __restrict__ o,
                             int rows, int Cx, int Cg, int Np, int ldo)
{
  long i = (long)blockIdx.x * blockDim.x + threadIdx.x;
  int Wt = 3 + Cx + Cg;
  long total = (long)rows * Wt;
  if (i >= total) return;
  int row = (int)(i / Wt), c = (int)(i % Wt);
  float v;
  if (c < 3)           v = p[(size_t)row * 3 + c];
  else if (c < 3 + Cx) v = x[(size_t)row * Cx + (c - 3)];
  else                 v = g[(size_t)(row / Np) * Cg + (c - 3 - Cx)];
  o[(size_t)row * ldo + c] = v;
}

// repack rows: dst[r*ldd + c] = src[r*w + c]  (alignment padding for K=515)
__global__ void k_pad_rows(const float* __restrict__ src, float* __restrict__ dst,
                           int w, int ldd, int total)
{
  int i = blockIdx.x * blockDim.x + threadIdx.x;
  if (i >= total) return;
  int r = i / w, c = i % w;
  dst[(size_t)r * ldd + c] = src[i];
}

// get_mirror: reflect points about plane defined by ab[b] = (A, B)
__global__ void k_mirror(const float* __restrict__ p, const float* __restrict__ ab,
                         float* __restrict__ o, int Bb, int Np)
{
  int i = blockIdx.x * blockDim.x + threadIdx.x;
  if (i >= Bb * Np) return;
  int b = i / Np;
  float A = ab[b * 2 + 0], Bc = ab[b * 2 + 1];
  float x = p[(size_t)i * 3 + 0], y = p[(size_t)i * 3 + 1], z = p[(size_t)i * 3 + 2];
  float t = 2.f * (A * x + Bc * z) / (A * A + Bc * Bc + 1e-8f);
  o[(size_t)i * 3 + 0] = x - A * t;
  o[(size_t)i * 3 + 1] = y;
  o[(size_t)i * 3 + 2] = z - Bc * t;
}

// dst[b, off+n] = src[b, n]   (points, 3 floats each)
__global__ void k_scatter_points(const float* __restrict__ src, float* __restrict__ dst,
                                 int NpSrc, int NpDst, int off, int total)
{
  int i = blockIdx.x * blockDim.x + threadIdx.x;
  if (i >= total) return;
  int k = i % 3, t = i / 3;
  int b = t / NpSrc, n = t % NpSrc;
  dst[((size_t)b * NpDst + off + n) * 3 + k] = src[i];
}

// ---------------------------------------------------------------------------
// Farthest point sampling: 1 workgroup (1024 threads = 32 waves) per batch;
// points + min-dist in LDS.  Per step: per-thread scan, wave32 shuffle argmax
// (ties -> lower index, matching jnp.argmax), cross-wave reduce in wave 0.
// Only 2 __syncthreads per step.  dynamic LDS: 4*NpIn floats.
// ---------------------------------------------------------------------------
__global__ __launch_bounds__(FPS_THREADS) void k_fps(const float* __restrict__ pts,
                                                     float* __restrict__ out,
                                                     int NpIn, int NOut)
{
  extern __shared__ float smem[];
  float* sx = smem;
  float* sy = sx + NpIn;
  float* sz = sy + NpIn;
  float* md = sz + NpIn;
  __shared__ float rwv[32];
  __shared__ int   rwi[32];
  __shared__ int   lastSh;

  const int tid  = threadIdx.x;
  const int lane = tid & 31;
  const int wv   = tid >> 5;
  int b = blockIdx.x;
  const float* P = pts + (size_t)b * NpIn * 3;
  float* O = out + (size_t)b * NOut * 3;

  for (int n = tid; n < NpIn; n += FPS_THREADS) {
    sx[n] = P[n * 3 + 0];
    sy[n] = P[n * 3 + 1];
    sz[n] = P[n * 3 + 2];
    md[n] = 1e10f;
  }
  __syncthreads();

  int last = 0;
  for (int s = 0; s < NOut; ++s) {
    float lx = sx[last], ly = sy[last], lz = sz[last];
    if (tid == 0) { O[s * 3] = lx; O[s * 3 + 1] = ly; O[s * 3 + 2] = lz; }
    float bv = -1.f; int bi = 0;
    for (int n = tid; n < NpIn; n += FPS_THREADS) {
      float dx = sx[n] - lx, dy = sy[n] - ly, dz = sz[n] - lz;
      float m = fminf(md[n], dx * dx + dy * dy + dz * dz);
      md[n] = m;
      if (m > bv) { bv = m; bi = n; }
    }
#pragma unroll
    for (int off = 16; off > 0; off >>= 1) {      // wave32 shuffle argmax
      float ov = __shfl_down(bv, off, 32);
      int   oi = __shfl_down(bi, off, 32);
      if (ov > bv || (ov == bv && oi < bi)) { bv = ov; bi = oi; }
    }
    if (lane == 0) { rwv[wv] = bv; rwi[wv] = bi; }
    __syncthreads();
    if (wv == 0) {
      bv = rwv[lane]; bi = rwi[lane];
#pragma unroll
      for (int off = 16; off > 0; off >>= 1) {
        float ov = __shfl_down(bv, off, 32);
        int   oi = __shfl_down(bi, off, 32);
        if (ov > bv || (ov == bv && oi < bi)) { bv = ov; bi = oi; }
      }
      if (lane == 0) lastSh = bi;
    }
    __syncthreads();
    last = lastSh;
  }
}

// ---------------------------------------------------------------------------
// Ball query (radius 0.25, nsample 32) + feature max.  One wave per query.
// ---------------------------------------------------------------------------
__global__ __launch_bounds__(THREADS) void k_qgm(const float* __restrict__ pts,
                                                 const float* __restrict__ feat,
                                                 float* __restrict__ out,
                                                 int Bb, int Np, int C)
{
  __shared__ int sidx[8][32];
  int wave = threadIdx.x >> 5, lane = threadIdx.x & 31;
  int q = blockIdx.x * 8 + wave;
  int total = Bb * Np;
  bool active = q < total;
  int b = active ? q / Np : 0;
  int qi = active ? q % Np : 0;
  const float* P = pts + (size_t)b * Np * 3;

  if (active && lane == 0) {
    float qx = P[qi * 3], qy = P[qi * 3 + 1], qz = P[qi * 3 + 2];
    int cnt = 0;
    for (int n = 0; n < Np && cnt < 32; ++n) {
      float dx = P[n * 3] - qx, dy = P[n * 3 + 1] - qy, dz = P[n * 3 + 2] - qz;
      if (dx * dx + dy * dy + dz * dz < 0.0625f) sidx[wave][cnt++] = n;
    }
    int pad = (cnt > 0) ? sidx[wave][0] : (Np - 1);
    for (int j = cnt; j < 32; ++j) sidx[wave][j] = pad;
  }
  __syncthreads();
  if (!active) return;

  for (int c = lane; c < C; c += 32) {
    float m = -3.4e38f;
#pragma unroll 4
    for (int j = 0; j < 32; ++j) {
      int n = sidx[wave][j];
      m = fmaxf(m, feat[((size_t)b * Np + n) * C + c]);
    }
    out[((size_t)b * Np + qi) * C + c] = m;
  }
}

// out[b, n*4+u, k] = P[b,n,k] + S[b*Np+n, u*3+k]   (UPN=4)
__global__ void k_shift_add(const float* __restrict__ P, const float* __restrict__ S,
                            float* __restrict__ out, int total)
{
  int i = blockIdx.x * blockDim.x + threadIdx.x;
  if (i >= total) return;
  int k = i % 3, t = i / 3;
  int u = t % 4, pnt = t / 4;
  out[i] = P[(size_t)pnt * 3 + k] + S[(size_t)pnt * 12 + u * 3 + k];
}

// ---------------------------------------------------------------------------
// host helpers
// ---------------------------------------------------------------------------
static inline void gemm_s(hipStream_t s, const float* X, const float* W, const float* b,
                          float* Y, int M, int K, int Co, int act, int ldx, int ldw)
{
  dim3 g((M + 511) / 512, (Co + 15) / 16);
  if ((M & 63) == 0)
    k_gemm4<<<g, THREADS, 0, s>>>(X, W, b, Y, M, K, Co, act, ldx, ldw);
  else
    k_gemm<<<g, THREADS, 0, s>>>(X, W, b, Y, M, K, Co, act, ldx, ldw);
}
static inline void gemm(hipStream_t s, const float* X, const float* W, const float* b,
                        float* Y, int M, int K, int Co, int act)
{
  gemm_s(s, X, W, b, Y, M, K, Co, act, K, K);
}
static inline void rowmax(hipStream_t s, const float* X, float* g, int Bb, int Np, int C)
{
  dim3 gr(Bb, (C + THREADS - 1) / THREADS);
  k_rowmax<<<gr, THREADS, 0, s>>>(X, g, Np, C);
}
static inline int nblk(long total) { return (int)((total + THREADS - 1) / THREADS); }

extern "C" void kernel_launch(void* const* d_in, const int* in_sizes, int n_in,
                              void* d_out_v, int out_size, void* d_ws, size_t ws_size,
                              hipStream_t stream)
{
  (void)in_sizes; (void)n_in; (void)out_size; (void)ws_size;
  const float* input_R = (const float*)d_in[0];
  const float* input_A = (const float*)d_in[1];
  auto Wp = [&](int i) { return (const float*)d_in[i]; };
  float* out = (float*)d_out_v;

  const int Bb = 16, N = 2048;
  const size_t PN3 = (size_t)Bb * N * 3;   // 98304

  // output offsets (return-tuple order, flat floats)
  float* o_point_R   = out + 0;
  float* o_point_R_3 = out + 24576;
  float* o_point_A   = out + 122880;
  float* o_point_A_3 = out + 147456;
  float* o_irpr0     = out + 245760;
  float* o_d_f_R     = out + 344064;
  float* o_d_f_A     = out + 344080;
  float* o_d_p_R     = out + 344096;
  float* o_d_p_R_3   = out + 344112;
  float* o_d_p_A     = out + 344128;
  float* o_point_R_0 = out + 344144;
  float* o_input_R_M = out + 368720;

  // ---- workspace bump allocator (sizes all multiples of 4 -> 16B bases) ----
  float* wsp = (float*)d_ws;
  auto alloc = [&](size_t n) { float* p = wsp; wsp += (n + 3) & ~(size_t)3; return p; };
  float* BIG0 = alloc((size_t)98304 * 512);
  float* BIG1 = alloc((size_t)98304 * 512);
  float* M0 = alloc((size_t)16384 * 512);
  float* M1 = alloc((size_t)16384 * 512);
  float* M2 = alloc((size_t)16384 * 512);
  float* M3 = alloc((size_t)16384 * 520);   // padded stride for K=515
  float* WR4 = alloc((size_t)512 * 520);    // repacked r4.W0
  float* DX  = alloc(32 * 1536);
  float* PR0 = alloc(16 * 1536);
  float* D0  = alloc(32 * 1536);
  float* D1  = alloc(32 * 1536);
  float* gbuf = alloc(48 * 512);
  float* fR0  = alloc(16 * 512);
  float* ab   = alloc(32);
  float* feat = alloc(32 * 512);
  float* xmr  = alloc(32 * 256);
  float* xm3  = alloc(32 * 256);
  float* featdp = alloc(48 * 256);
  float* PM   = alloc(PN3);
  float* PC   = alloc((size_t)Bb * 4608 * 3);
  float* PF   = alloc(PN3);
  float* PX   = alloc(2 * PN3);
  float* X3   = alloc(2 * PN3);
  float* PM3  = alloc(PN3);
  float* PC3  = alloc((size_t)Bb * 4096 * 3);
  float* PR3f = alloc(PN3);
  float* PD   = alloc(3 * PN3);
  float* dfout = alloc(64);
  float* dpout = alloc(64);

  auto cpy = [&](float* dst, const float* src, size_t n) {
    hipMemcpyAsync(dst, src, n * sizeof(float), hipMemcpyDeviceToDevice, stream);
  };

  // PCN encoder: mlp1 3->128(relu)->256, global max, concat, mlp2 512->512(relu)->Co, max
  auto encoder = [&](const float* X, int rows, int Np, int i10, int i11,
                     int i20, int i21, int coutF, float* featOut) {
    gemm(stream, X,    Wp(i10), Wp(i10 + 1), BIG0, rows, 3,   128, 1);
    gemm(stream, BIG0, Wp(i11), Wp(i11 + 1), BIG1, rows, 128, 256, 0);
    rowmax(stream, BIG1, gbuf, rows / Np, Np, 256);
    k_concat_fg<<<nblk((long)rows * 512), THREADS, 0, stream>>>(BIG1, gbuf, BIG0, rows, 256, Np);
    gemm(stream, BIG0, Wp(i20), Wp(i20 + 1), BIG1, rows, 512, 512, 1);
    gemm(stream, BIG1, Wp(i21), Wp(i21 + 1), BIG0, rows, 512, coutF, 0);
    rowmax(stream, BIG0, featOut, rows / Np, Np, coutF);
  };
  // 5-layer decoder 512->512->512->1024->1024->1536 (relu x4)
  auto mlp5 = [&](const float* X, int rows, int base, float* outBuf) {
    gemm(stream, X,  Wp(base + 0), Wp(base + 1), D0, rows, 512,  512,  1);
    gemm(stream, D0, Wp(base + 2), Wp(base + 3), D1, rows, 512,  512,  1);
    gemm(stream, D1, Wp(base + 4), Wp(base + 5), D0, rows, 512,  1024, 1);
    gemm(stream, D0, Wp(base + 6), Wp(base + 7), D1, rows, 1024, 1024, 1);
    gemm(stream, D1, Wp(base + 8), Wp(base + 9), outBuf, rows, 1024, 1536, 0);
  };

  // repack r4.W0 (512 x 515) into 520-stride buffer (alignment for fast path)
  k_pad_rows<<<nblk((long)512 * 515), THREADS, 0, stream>>>(Wp(82), WR4, 515, 520, 512 * 515);

  // ---------------- E_R encoder on input_R -> f_R_0 ----------------
  encoder(input_R, Bb * N, N, 30, 32, 34, 36, 512, fR0);

  // ---------------- D_R decode -> point_R_0 ----------------
  mlp5(fR0, 16, 12, PR0);
  cpy(o_point_R_0, PR0, 24576);

  // ---------------- mirror plane: 512->128->128->2, tanh ----------------
  gemm(stream, fR0, Wp(58), Wp(59), D0, 16, 512, 128, 1);
  gemm(stream, D0,  Wp(60), Wp(61), D1, 16, 128, 128, 1);
  gemm(stream, D1,  Wp(62), Wp(63), ab, 16, 128, 2,   2);

  k_mirror<<<nblk(Bb * N), THREADS, 0, stream>>>(input_R, ab, PM, Bb, N);
  cpy(o_input_R_M, PM, PN3);

  // irpr0_raw = [input_R | mirror | point_R_0] per batch, then FPS -> 2048
  k_scatter_points<<<nblk(PN3), THREADS, 0, stream>>>(input_R, PC, 2048, 4608, 0, (int)PN3);
  k_scatter_points<<<nblk(PN3), THREADS, 0, stream>>>(PM,      PC, 2048, 4608, 2048, (int)PN3);
  k_scatter_points<<<nblk(Bb * 512 * 3), THREADS, 0, stream>>>(PR0, PC, 512, 4608, 4096, Bb * 512 * 3);
  k_fps<<<Bb, FPS_THREADS, 4 * 4608 * sizeof(float), stream>>>(PC, PF, 4608, 2048);
  cpy(o_irpr0, PF, PN3);

  // ---------------- E_A encoder on [irpr0; input_A] -> feat (2B,512) --------
  cpy(PX, PF, PN3);
  cpy(PX + PN3, input_A, PN3);
  encoder(PX, 2 * Bb * N, N, 22, 24, 26, 28, 512, feat);

  // ---------------- D_A decode -> points (2B,512,3) ----------------
  mlp5(feat, 32, 2, DX);
  cpy(o_point_R, DX, 24576);
  cpy(o_point_A, DX + 24576, 24576);

  // ---------------- upsampling_refine ----------------
  const int RR = 32 * 512;  // 16384 rows
  gemm(stream, DX, Wp(64), Wp(65), M0, RR, 3,   256, 1);
  gemm(stream, M0, Wp(66), Wp(67), M1, RR, 256, 256, 1);
  gemm(stream, M1, Wp(68), Wp(69), M0, RR, 256, 256, 0);
  rowmax(stream, M0, xmr, 32, 512, 256);
  k_concat_fg<<<nblk((long)RR * 512), THREADS, 0, stream>>>(M0, xmr, M1, RR, 256, 512);
  gemm(stream, M1, Wp(70), Wp(71), M2, RR, 512, 256, 1);
  gemm(stream, M2, Wp(72), Wp(73), M0, RR, 256, 256, 1);
  gemm(stream, M0, Wp(74), Wp(75), M2, RR, 256, 256, 0);     // x -> M2
  k_qgm<<<(RR + 7) / 8, THREADS, 0, stream>>>(DX, M2, M0, 32, 512, 256);  // xl -> M0
  k_concat2<<<nblk((long)RR * 512), THREADS, 0, stream>>>(M2, M0, M1, RR, 256, 256);
  gemm(stream, M1, Wp(76), Wp(77), M0, RR, 512, 256, 1);
  gemm(stream, M0, Wp(78), Wp(79), M2, RR, 256, 256, 1);
  gemm(stream, M2, Wp(80), Wp(81), M0, RR, 256, 256, 0);     // x3f -> M0
  rowmax(stream, M0, xm3, 32, 512, 256);
  k_concat_pxg<<<nblk((long)RR * 515), THREADS, 0, stream>>>(DX, M0, xm3, M3, RR, 256, 256, 512, 520);
  gemm_s(stream, M3, WR4, Wp(83), M1, RR, 515, 512, 1, 520, 520);
  gemm(stream, M1, Wp(84), Wp(85), M2, RR, 512, 512, 1);
  gemm(stream, M2, Wp(86), Wp(87), M1, RR, 512, 12, 0);      // shifts -> M1
  k_shift_add<<<nblk(2 * (long)PN3), THREADS, 0, stream>>>(DX, M1, X3, (int)(2 * PN3));
  cpy(o_point_A_3, X3 + PN3, PN3);

  // ---------------- point_R_3 = fps([x3_R | mirror(x3_R)]) ----------------
  k_mirror<<<nblk(Bb * N), THREADS, 0, stream>>>(X3, ab, PM3, Bb, N);
  k_scatter_points<<<nblk(PN3), THREADS, 0, stream>>>(X3,  PC3, 2048, 4096, 0,    (int)PN3);
  k_scatter_points<<<nblk(PN3), THREADS, 0, stream>>>(PM3, PC3, 2048, 4096, 2048, (int)PN3);
  k_fps<<<Bb, FPS_THREADS, 4 * 4096 * sizeof(float), stream>>>(PC3, PR3f, 4096, 2048);
  cpy(o_point_R_3, PR3f, PN3);

  // ---------------- d_f head: 512->64->64->1, sigmoid ----------------
  gemm(stream, feat, Wp(38), Wp(39), D0, 32, 512, 64, 1);
  gemm(stream, D0,   Wp(40), Wp(41), D1, 32, 64,  64, 1);
  gemm(stream, D1,   Wp(42), Wp(43), dfout, 32, 64, 1, 3);
  cpy(o_d_f_R, dfout, 16);
  cpy(o_d_f_A, dfout + 16, 16);

  // ---------------- d_p head: dp_enc on [irpr0; point_R_3; input_A] ---------
  cpy(PD, PF, PN3);
  cpy(PD + PN3, PR3f, PN3);
  cpy(PD + 2 * PN3, input_A, PN3);
  encoder(PD, 3 * Bb * N, N, 44, 46, 48, 50, 256, featdp);
  gemm(stream, featdp, Wp(52), Wp(53), D0, 48, 256, 64, 1);
  gemm(stream, D0,     Wp(54), Wp(55), D1, 48, 64,  64, 1);
  gemm(stream, D1,     Wp(56), Wp(57), dpout, 48, 64, 1, 3);
  cpy(o_d_p_R,   dpout, 16);
  cpy(o_d_p_R_3, dpout + 16, 16);
  cpy(o_d_p_A,   dpout + 32, 16);
}
